// TLSTM_7198365188123
// MI455X (gfx1250) — compile-verified, matrix operand order checked by reading, not run on hardware
//
#include <hip/hip_runtime.h>
#include <hip/hip_bf16.h>

// ---------------------------------------------------------------------------
// TLSTM on MI455X (gfx1250): bf16 WMMA, fused per-step kernel.
// v2: stage A-rows once (no barriers in K-loop), scalar (SGPR) B-tile bases
// via readfirstlane, hoisted operand loads, conflict-free LDS pitch.
// ---------------------------------------------------------------------------

typedef __attribute__((ext_vector_type(16))) __bf16 v16bf;
typedef __attribute__((ext_vector_type(8)))  float  v8f;

static constexpr int Bb = 128;   // batch
static constexpr int Ss = 512;   // seq len
static constexpr int Dd = 256;   // input dim
static constexpr int Hh = 512;   // hidden
static constexpr int Oo = 256;   // out dim

static constexpr int TILE = 512;       // bf16 elems per swizzled B tile (32x16)
static constexpr int PH = 516;         // LDS row pitch for 512-wide staging
static constexpr int PX = 260;         // LDS row pitch for 256-wide staging

#define WMMA_BF16(A, B, C)                                                   \
  __builtin_amdgcn_wmma_f32_16x16x32_bf16(false, (A), false, (B),            \
                                          (short)0, (C), false, false)

__device__ __forceinline__ float sigm(float x) {
  return 1.0f / (1.0f + __expf(-x));
}

// A-operand (16x32 MxK, bf16) per-lane gather from staged LDS rows.
// lane<16 : e0..7 -> K0..7,  e8..15 -> K16..23   (M = lane)
// lane>=16: e0..7 -> K8..15, e8..15 -> K24..31   (M = lane-16)
// Batched as 4 x b128 LDS loads, then packed converts.
__device__ __forceinline__ v16bf gatherA(const float* __restrict__ lds,
                                         int lane, int kt, int pitch) {
  const int m  = lane & 15;
  const int kb = (lane >= 16) ? 8 : 0;
  const float* p = lds + m * pitch + kt * 32 + kb;
  float4 x0 = *(const float4*)(p);
  float4 x1 = *(const float4*)(p + 4);
  float4 x2 = *(const float4*)(p + 16);
  float4 x3 = *(const float4*)(p + 20);
  v16bf a;
  a[0]  = (__bf16)x0.x; a[1]  = (__bf16)x0.y; a[2]  = (__bf16)x0.z; a[3]  = (__bf16)x0.w;
  a[4]  = (__bf16)x1.x; a[5]  = (__bf16)x1.y; a[6]  = (__bf16)x1.z; a[7]  = (__bf16)x1.w;
  a[8]  = (__bf16)x2.x; a[9]  = (__bf16)x2.y; a[10] = (__bf16)x2.z; a[11] = (__bf16)x2.w;
  a[12] = (__bf16)x3.x; a[13] = (__bf16)x3.y; a[14] = (__bf16)x3.z; a[15] = (__bf16)x3.w;
  return a;
}

// Pre-swizzled B tile: 512 bf16 laid out lane-major, 16 per lane (32B load).
__device__ __forceinline__ v16bf loadB(const __bf16* __restrict__ tile,
                                       int lane) {
  return *(const v16bf*)(tile + lane * 16);
}

// ---------------------------------------------------------------------------
// Weight pre-swizzle: W[K][N] fp32 row-major -> bf16 tiles (nt, kt):
//   out[(nt*KT + kt)*512 + lane*16 + e] = W[(kt*32 + Kmap(lane,e))*N + nt*16 + lane%16]
// ---------------------------------------------------------------------------
__global__ void swizzle_weight_kernel(const float* __restrict__ W,
                                      __bf16* __restrict__ out, int K, int N) {
  int idx = blockIdx.x * blockDim.x + threadIdx.x;
  int total = K * N;
  if (idx >= total) return;
  int e    = idx & 15;
  int l    = (idx >> 4) & 31;
  int tile = idx >> 9;
  int ktiles = K >> 5;
  int kt = tile % ktiles;
  int nt = tile / ktiles;
  int kk = e + ((e >= 8) ? 8 : 0) + ((l >= 16) ? 8 : 0);
  int krow = kt * 32 + kk;
  int ncol = nt * 16 + (l & 15);
  out[idx] = (__bf16)W[(size_t)krow * N + ncol];
}

__global__ void zero_f32_kernel(float* __restrict__ p, int n) {
  int i = blockIdx.x * blockDim.x + threadIdx.x;
  if (i < n) p[i] = 0.0f;
}

// ---------------------------------------------------------------------------
// One fused timestep. Grid (8,8): blockIdx.y = b-tile (16 rows),
// blockIdx.x = group of 4 h-tiles; 128 threads = 4 waves, wave w owns h-tile
// blockIdx.x*4+w. Full A rows (h, c, x) staged in LDS ONCE; K-loops are
// barrier-free so the scheduler pipelines B loads over WMMAs.
// ---------------------------------------------------------------------------
__global__ __launch_bounds__(128) void tlstm_step_kernel(
    const float* __restrict__ inputs,      // [B,S,D]
    const float* __restrict__ time_diffs,  // [B,S]
    const __bf16* __restrict__ Usw,        // swizzled [H x 4H]
    const __bf16* __restrict__ Wallsw,     // swizzled [D x 4H]
    const __bf16* __restrict__ Wdsw,       // swizzled [H x H]
    const float* __restrict__ b_all, const float* __restrict__ b_u,
    const float* __restrict__ b_d,
    const float* __restrict__ hprev_base, long long hprev_stride,
    const float* __restrict__ cprev_base, long long cprev_stride,
    float* __restrict__ hs,                // [B,S,H] region of d_out
    float* __restrict__ cs,                // [B,S,H] region of d_out
    int t) {
  __shared__ __align__(16) float AsH[16 * PH];  // h_prev rows, 33 KB
  __shared__ __align__(16) float AsC[16 * PH];  // c_prev rows, 33 KB
  __shared__ __align__(16) float AsX[16 * PX];  // x_t rows,    16.6 KB

  const int tid  = threadIdx.x;
  const int lane = tid & 31;
  const int b0   = blockIdx.y * 16;
  // Wave-uniform h-tile index forced into an SGPR -> scalar B-tile bases.
  const int jts = __builtin_amdgcn_readfirstlane(
      (int)(blockIdx.x * 4 + (threadIdx.x >> 5)));
  const int hj = jts * 16;

  constexpr int KT_H = Hh / 32;  // 16
  constexpr int KT_D = Dd / 32;  // 8
  constexpr int NT_H = Hh / 16;  // 32 (gate stride in 4H tile space)

  // ---- Stage full A rows (coalesced), one barrier total ----
#pragma unroll
  for (int j = 0; j < 16; ++j) {            // 16x512 of h and c
    int idx4 = tid + j * 128;
    int row  = idx4 >> 7;
    int col  = (idx4 & 127) * 4;
    const float* ph = hprev_base + (size_t)(b0 + row) * hprev_stride + col;
    const float* pc = cprev_base + (size_t)(b0 + row) * cprev_stride + col;
    *(float4*)&AsH[row * PH + col] = *(const float4*)ph;
    *(float4*)&AsC[row * PH + col] = *(const float4*)pc;
  }
#pragma unroll
  for (int j = 0; j < 8; ++j) {             // 16x256 of x_t
    int idx4 = tid + j * 128;
    int row  = idx4 >> 6;
    int col  = (idx4 & 63) * 4;
    const float* px = inputs + ((size_t)(b0 + row) * Ss + t) * Dd + col;
    *(float4*)&AsX[row * PX + col] = *(const float4*)px;
  }
  __syncthreads();

  // Scalar (SGPR) base pointers for the 5 B-operand streams.
  const __bf16* baseF = Usw  + (size_t)(jts)            * (TILE * KT_H);
  const __bf16* baseI = Usw  + (size_t)(NT_H + jts)     * (TILE * KT_H);
  const __bf16* baseO = Usw  + (size_t)(2 * NT_H + jts) * (TILE * KT_H);
  const __bf16* baseG = Usw  + (size_t)(3 * NT_H + jts) * (TILE * KT_H);
  const __bf16* baseC = Wdsw + (size_t)(jts)            * (TILE * KT_H);
  const __bf16* xbF = Wallsw + (size_t)(jts)            * (TILE * KT_D);
  const __bf16* xbI = Wallsw + (size_t)(NT_H + jts)     * (TILE * KT_D);
  const __bf16* xbO = Wallsw + (size_t)(2 * NT_H + jts) * (TILE * KT_D);
  const __bf16* xbG = Wallsw + (size_t)(3 * NT_H + jts) * (TILE * KT_D);
  __builtin_prefetch((const void*)baseF, 0, 1);   // global_prefetch_b8
  __builtin_prefetch((const void*)baseC, 0, 1);

  v8f accF = {}, accI = {}, accO = {}, accG = {}, accC = {};

  // ---- K-loop over hidden: h@U (4 gates) + c@W_d, barrier-free ----
#pragma unroll
  for (int kt = 0; kt < KT_H; ++kt) {
    v16bf bF = loadB(baseF + (size_t)kt * TILE, lane);
    v16bf bI = loadB(baseI + (size_t)kt * TILE, lane);
    v16bf bO = loadB(baseO + (size_t)kt * TILE, lane);
    v16bf bG = loadB(baseG + (size_t)kt * TILE, lane);
    v16bf bC = loadB(baseC + (size_t)kt * TILE, lane);
    v16bf aH = gatherA(AsH, lane, kt, PH);
    v16bf aC = gatherA(AsC, lane, kt, PH);
    accF = WMMA_BF16(aH, bF, accF);
    accI = WMMA_BF16(aH, bI, accI);
    accO = WMMA_BF16(aH, bO, accO);
    accG = WMMA_BF16(aH, bG, accG);
    accC = WMMA_BF16(aC, bC, accC);
  }

  // ---- K-loop over input dim: x_t @ W_all ----
#pragma unroll
  for (int kt = 0; kt < KT_D; ++kt) {
    v16bf bF = loadB(xbF + (size_t)kt * TILE, lane);
    v16bf bI = loadB(xbI + (size_t)kt * TILE, lane);
    v16bf bO = loadB(xbO + (size_t)kt * TILE, lane);
    v16bf bG = loadB(xbG + (size_t)kt * TILE, lane);
    v16bf aX = gatherA(AsX, lane, kt, PX);
    accF = WMMA_BF16(aX, bF, accF);
    accI = WMMA_BF16(aX, bI, accI);
    accO = WMMA_BF16(aX, bO, accO);
    accG = WMMA_BF16(aX, bG, accG);
  }

  // ---- Gates + state update (C/D layout: VGPR r -> M = r + (lane>=16)*8) ----
  const int n     = lane & 15;
  const int rbase = (lane >= 16) ? 8 : 0;
  const float biasF = b_all[jts * 16 + n]                + b_u[jts * 16 + n];
  const float biasI = b_all[(NT_H + jts) * 16 + n]       + b_u[(NT_H + jts) * 16 + n];
  const float biasO = b_all[(2 * NT_H + jts) * 16 + n]   + b_u[(2 * NT_H + jts) * 16 + n];
  const float biasG = b_all[(3 * NT_H + jts) * 16 + n]   + b_u[(3 * NT_H + jts) * 16 + n];
  const float biasC = b_d[hj + n];

#pragma unroll
  for (int r = 0; r < 8; ++r) {
    const int row = b0 + rbase + r;
    const float f   = sigm(accF[r] + biasF);
    const float ig  = sigm(accI[r] + biasI);
    const float og  = sigm(accO[r] + biasO);
    const float g   = tanhf(accG[r] + biasG);
    const float cst = sigm(accC[r] + biasC);
    const float cprev = AsC[(rbase + r) * PH + hj + n];  // from staged LDS
    const float td    = time_diffs[(size_t)row * Ss + t];
    const float cadj  = cprev - cst + cst * td;
    const float ct = f * cadj + ig * g;
    const float ht = og * tanhf(ct);
    const size_t o = ((size_t)row * Ss + t) * Hh + hj + n;
    hs[o] = ht;
    cs[o] = ct;
  }
}

// ---------------------------------------------------------------------------
// outputs = hs @ W_out + b_out   (M = B*S = 65536, K = H = 512, N = O = 256)
// 4096 blocks x 128 threads; block stages one 16x512 hs row-panel once,
// 4 waves cover the 16 output-column tiles (4 each), barrier-free K-loop.
// ---------------------------------------------------------------------------
__global__ __launch_bounds__(128) void out_proj_kernel(
    const float* __restrict__ hs, const __bf16* __restrict__ Woutsw,
    const float* __restrict__ b_out, float* __restrict__ out) {
  __shared__ __align__(16) float As[16 * PH];
  const int tid  = threadIdx.x;
  const int lane = tid & 31;
  const int wv = __builtin_amdgcn_readfirstlane((int)(threadIdx.x >> 5));
  const int m0 = blockIdx.x * 16;
  constexpr int KT = Hh / 32;  // 16

#pragma unroll
  for (int j = 0; j < 16; ++j) {
    int idx4 = tid + j * 128;
    int row  = idx4 >> 7;
    int col  = (idx4 & 127) * 4;
    *(float4*)&As[row * PH + col] =
        *(const float4*)&hs[(size_t)(m0 + row) * Hh + col];
  }
  __syncthreads();

  const __bf16* base0 = Woutsw + (size_t)(wv * 4 + 0) * (TILE * KT);
  const __bf16* base1 = Woutsw + (size_t)(wv * 4 + 1) * (TILE * KT);
  const __bf16* base2 = Woutsw + (size_t)(wv * 4 + 2) * (TILE * KT);
  const __bf16* base3 = Woutsw + (size_t)(wv * 4 + 3) * (TILE * KT);

  v8f acc0 = {}, acc1 = {}, acc2 = {}, acc3 = {};
#pragma unroll
  for (int kt = 0; kt < KT; ++kt) {
    v16bf b0 = loadB(base0 + (size_t)kt * TILE, lane);
    v16bf b1 = loadB(base1 + (size_t)kt * TILE, lane);
    v16bf b2 = loadB(base2 + (size_t)kt * TILE, lane);
    v16bf b3 = loadB(base3 + (size_t)kt * TILE, lane);
    v16bf a  = gatherA(As, lane, kt, PH);
    acc0 = WMMA_BF16(a, b0, acc0);
    acc1 = WMMA_BF16(a, b1, acc1);
    acc2 = WMMA_BF16(a, b2, acc2);
    acc3 = WMMA_BF16(a, b3, acc3);
  }

  const int n     = lane & 15;
  const int rbase = (lane >= 16) ? 8 : 0;
  const v8f* accs[4] = {&acc0, &acc1, &acc2, &acc3};
#pragma unroll
  for (int it = 0; it < 4; ++it) {
    const int nt = wv * 4 + it;
    const float bias = b_out[nt * 16 + n];
#pragma unroll
    for (int r = 0; r < 8; ++r) {
      out[(size_t)(m0 + rbase + r) * Oo + nt * 16 + n] = (*accs[it])[r] + bias;
    }
  }
}

// ---------------------------------------------------------------------------
extern "C" void kernel_launch(void* const* d_in, const int* in_sizes, int n_in,
                              void* d_out, int out_size, void* d_ws,
                              size_t ws_size, hipStream_t stream) {
  (void)in_sizes; (void)n_in; (void)out_size; (void)ws_size;

  const float* inputs  = (const float*)d_in[0];
  const float* tdiffs  = (const float*)d_in[1];
  // d_in[2] = seq_lens (unused by the reference computation)
  const float* W_all   = (const float*)d_in[3];
  const float* b_all   = (const float*)d_in[4];
  const float* U_all   = (const float*)d_in[5];
  const float* b_u     = (const float*)d_in[6];
  const float* W_d     = (const float*)d_in[7];
  const float* b_d     = (const float*)d_in[8];
  const float* W_out   = (const float*)d_in[9];
  const float* b_out   = (const float*)d_in[10];

  float* out = (float*)d_out;                    // [B,S,O]
  float* hs  = out + (size_t)Bb * Ss * Oo;       // [B,S,H]
  float* cs  = hs + (size_t)Bb * Ss * Hh;        // [B,S,H]

  // Workspace layout (~4.25 MB): bf16-swizzled weights + zero state.
  char* ws = (char*)d_ws;
  __bf16* Usw    = (__bf16*)(ws);                              // 2 MB
  __bf16* Wallsw = (__bf16*)(ws + (2u << 20));                 // 1 MB
  __bf16* Wdsw   = (__bf16*)(ws + (3u << 20));                 // 512 KB
  __bf16* Woutsw = (__bf16*)(ws + (3u << 20) + (512u << 10));  // 256 KB
  float*  zbuf   = (float*)(ws + (4u << 20));                  // 256 KB

  auto sw = [&](const float* W, __bf16* o, int K, int N) {
    int total = K * N;
    swizzle_weight_kernel<<<(total + 255) / 256, 256, 0, stream>>>(W, o, K, N);
  };
  sw(U_all, Usw,    Hh, 4 * Hh);
  sw(W_all, Wallsw, Dd, 4 * Hh);
  sw(W_d,   Wdsw,   Hh, Hh);
  sw(W_out, Woutsw, Hh, Oo);
  zero_f32_kernel<<<(Bb * Hh + 255) / 256, 256, 0, stream>>>(zbuf, Bb * Hh);

  // Serial recurrence: one fused kernel per timestep.
  dim3 grid(8, 8);  // (h-tile groups, b-tiles)
  for (int t = 0; t < Ss; ++t) {
    const float* hp = t ? (hs + (size_t)(t - 1) * Hh) : zbuf;
    const float* cp = t ? (cs + (size_t)(t - 1) * Hh) : zbuf;
    long long str = t ? (long long)Ss * Hh : (long long)Hh;
    tlstm_step_kernel<<<grid, 128, 0, stream>>>(
        inputs, tdiffs, Usw, Wallsw, Wdsw, b_all, b_u, b_d,
        hp, str, cp, str, hs, cs, t);
  }

  // Output projection over all (b, t) rows.
  out_proj_kernel<<<(Bb * Ss) / 16, 128, 0, stream>>>(hs, Woutsw, b_out, out);
}
